// TFMobileBertSelfAttention_22058952032421
// MI455X (gfx1250) — compile-verified, matrix operand order checked by reading, not run on hardware
//
#include <hip/hip_runtime.h>

#define BB  8
#define SS  2048
#define HH  4
#define DHD 32
#define DD  128

typedef __attribute__((ext_vector_type(16))) _Float16 v16h;
typedef __attribute__((ext_vector_type(8)))  float    v8f;

// CDNA5 16-bit A/B fragment mapping (wave32):
// lane L holds row (A) / col (B) = L%16; element e -> K = (e&7) + 8*(L/16) + 16*(e>>3)
__device__ __forceinline__ int kmap(int e, int half) {
  return (e & 7) + (half << 3) + ((e >> 3) << 4);
}

// Load fragment from f16 matrix where element (row, k) = base[row*ld + k].
// Works for A (row = M row) and B (row = N column of a K-major view).
__device__ __forceinline__ v16h frag_f16(const _Float16* base, int ld, int row, int half) {
  const _Float16* p = base + (size_t)row * ld;
  v16h f;
#pragma unroll
  for (int e = 0; e < 16; e += 2) {
    int k = kmap(e, half);
    f[e]     = p[k];
    f[e + 1] = p[k + 1];
  }
  return f;
}

// A fragment from fp32 row-major source, converting to f16.
__device__ __forceinline__ v16h fragA_f32(const float* base, int ld, int row, int kc, int half) {
  const float* p = base + (size_t)row * ld + kc;
  v16h f;
#pragma unroll
  for (int e = 0; e < 16; e += 2) {
    int k = kmap(e, half);
    f[e]     = (_Float16)p[k];
    f[e + 1] = (_Float16)p[k + 1];
  }
  return f;
}

// B fragment from fp32 row-major weight W[k][col] (k strided by ld).
__device__ __forceinline__ v16h fragB_f32(const float* W, int ld, int kc, int col, int half) {
  v16h f;
#pragma unroll
  for (int e = 0; e < 16; e += 2) {
    int k = kc + kmap(e, half);
    f[e]     = (_Float16)W[(size_t)k * ld + col];
    f[e + 1] = (_Float16)W[(size_t)(k + 1) * ld + col];
  }
  return f;
}

__device__ __forceinline__ v8f wmma_f16(v16h a, v16h b, v8f c) {
  return __builtin_amdgcn_wmma_f32_16x16x32_f16(false, a, false, b, (short)0, c, false, false);
}

// ---------------- QKV projection: X(16-row tile) @ W + b, split heads ----------------
// grid = (B*S)/16 blocks, 256 threads (8 waves). Wave w owns column tile w (16 cols),
// loops over Q/K/V. Q,K stored (B,H,S,DH) f16; V stored transposed (B,H,DH,S) f16.
__global__ __launch_bounds__(256)
void proj_kernel(const float* __restrict__ qx, const float* __restrict__ kx,
                 const float* __restrict__ vx,
                 const float* __restrict__ Wq, const float* __restrict__ bq,
                 const float* __restrict__ Wk, const float* __restrict__ bk,
                 const float* __restrict__ Wv, const float* __restrict__ bv,
                 _Float16* __restrict__ qws, _Float16* __restrict__ kws,
                 _Float16* __restrict__ vtws) {
  int lane = threadIdx.x & 31;
  int wave = threadIdx.x >> 5;
  int half = lane >> 4;
  int r    = lane & 15;
  int m0   = blockIdx.x * 16;
  int col  = wave * 16 + r;     // output column 0..127
  int h    = col >> 5;
  int dh   = col & 31;

#pragma unroll
  for (int mat = 0; mat < 3; ++mat) {
    const float* X    = (mat == 0) ? qx : (mat == 1) ? kx : vx;
    const float* W    = (mat == 0) ? Wq : (mat == 1) ? Wk : Wv;
    const float* bias = (mat == 0) ? bq : (mat == 1) ? bk : bv;

    v8f c = {};
#pragma unroll
    for (int kc = 0; kc < DD; kc += 32) {
      v16h a = fragA_f32(X, DD, m0 + r, kc, half);
      v16h b = fragB_f32(W, DD, kc, col, half);
      c = wmma_f16(a, b, c);
    }
    float bias_v = bias[col];
#pragma unroll
    for (int v = 0; v < 8; ++v) {
      int row = m0 + v + 8 * half;          // C layout: VGPR v, half -> M = v + 8*half
      int bb  = row >> 11;                  // row / 2048
      int s   = row & (SS - 1);
      float val = c[v] + bias_v;
      if (mat == 2) {
        vtws[(((size_t)bb * HH + h) * DHD + dh) * SS + s] = (_Float16)val;
      } else {
        _Float16* dst = (mat == 0) ? qws : kws;
        dst[(((size_t)bb * HH + h) * SS + s) * DHD + dh] = (_Float16)val;
      }
    }
  }
}

// ---------------- fused flash attention: one wave per 16-query tile ----------------
// grid = B*H*(S/16)/8 blocks, 256 threads (8 waves, independent tiles).
__global__ __launch_bounds__(256)
void attn_kernel(const _Float16* __restrict__ qws, const _Float16* __restrict__ kws,
                 const _Float16* __restrict__ vtws, const float* __restrict__ amask,
                 const float* __restrict__ hmask, float* __restrict__ out) {
  __shared__ _Float16 Plds[8][16 * 32];     // per-wave P staging (C-layout -> A-layout)

  int lane = threadIdx.x & 31;
  int wave = threadIdx.x >> 5;
  int half = lane >> 4;
  int cn   = lane & 15;

  int g   = blockIdx.x * 8 + wave;          // tile id 0..4095
  int b   = g >> 9;                         // / (H * S/16) = 512
  int rem = g & 511;
  int h   = rem >> 7;                       // / 128
  int q0  = (rem & 127) * 16;

  const _Float16* qb  = qws  + (((size_t)b * HH + h) * SS) * DHD;
  const _Float16* kb  = kws  + (((size_t)b * HH + h) * SS) * DHD;
  const _Float16* vtb = vtws + (((size_t)b * HH + h) * DHD) * SS;
  const float*    mb  = amask + (size_t)b * SS;

  v16h aQ = frag_f16(qb, DHD, q0 + cn, half);   // 16x32 Q tile, K = DH = 32

  v8f acc0 = {}, acc1 = {};
  float m[8], l[8];
#pragma unroll
  for (int v = 0; v < 8; ++v) { m[v] = -3.0e38f; l[v] = 0.0f; }

  const float scale = 0.17677669529663687f;     // 1/sqrt(32)
  _Float16* P = &Plds[wave][0];
  v8f zero = {};

  for (int j = 0; j < SS; j += 32) {
    // scores: Q @ K^T for keys [j, j+32)
    v16h bK0 = frag_f16(kb, DHD, j + cn,      half);
    v16h bK1 = frag_f16(kb, DHD, j + 16 + cn, half);
    v8f s1 = wmma_f16(aQ, bK0, zero);
    v8f s2 = wmma_f16(aQ, bK1, zero);
    float mk0 = mb[j + cn];
    float mk1 = mb[j + 16 + cn];

#pragma unroll
    for (int v = 0; v < 8; ++v) {
      float x1 = s1[v] * scale + mk0;
      float x2 = s2[v] * scale + mk1;
      // row max over 32 keys: local max then butterfly across the 16-lane half
      float t = fmaxf(x1, x2);
#pragma unroll
      for (int o = 1; o < 16; o <<= 1) t = fmaxf(t, __shfl_xor(t, o, 32));
      float mn = fmaxf(m[v], t);
      float e1 = __expf(x1 - mn);
      float e2 = __expf(x2 - mn);
      float rs = e1 + e2;
#pragma unroll
      for (int o = 1; o < 16; o <<= 1) rs += __shfl_xor(rs, o, 32);
      float corr = __expf(m[v] - mn);
      l[v] = l[v] * corr + rs;
      acc0[v] *= corr;
      acc1[v] *= corr;
      m[v] = mn;
      // stage P (C-layout element (row = v+8*half, key = cn / 16+cn)) as f16
      P[(v + 8 * half) * 32 + cn]      = (_Float16)e1;
      P[(v + 8 * half) * 32 + 16 + cn] = (_Float16)e2;
    }

    asm volatile("s_wait_dscnt 0" ::: "memory");

    // ctx += P(16x32) @ V(32x32): A from LDS, B from transposed V (contiguous K-pairs)
    v16h aP  = frag_f16(P, 32, cn, half);
    v16h bV0 = frag_f16(vtb + j, SS, cn,      half);
    v16h bV1 = frag_f16(vtb + j, SS, 16 + cn, half);
    acc0 = wmma_f16(aP, bV0, acc0);
    acc1 = wmma_f16(aP, bV1, acc1);
  }

  float hm = hmask[h];
#pragma unroll
  for (int v = 0; v < 8; ++v) {
    int row = q0 + v + 8 * half;
    float inv = hm / l[v];
    size_t o = (((size_t)b * SS + row) * DD) + (size_t)h * DHD;
    out[o + cn]      = acc0[v] * inv;
    out[o + 16 + cn] = acc1[v] * inv;
  }
}

extern "C" void kernel_launch(void* const* d_in, const int* in_sizes, int n_in,
                              void* d_out, int out_size, void* d_ws, size_t ws_size,
                              hipStream_t stream) {
  (void)in_sizes; (void)n_in; (void)out_size; (void)ws_size;
  const float* qx    = (const float*)d_in[0];
  const float* kx    = (const float*)d_in[1];
  const float* vx    = (const float*)d_in[2];
  const float* amask = (const float*)d_in[3];
  const float* hmask = (const float*)d_in[4];
  const float* Wq    = (const float*)d_in[5];
  const float* bq    = (const float*)d_in[6];
  const float* Wk    = (const float*)d_in[7];
  const float* bk    = (const float*)d_in[8];
  const float* Wv    = (const float*)d_in[9];
  const float* bv    = (const float*)d_in[10];

  size_t n = (size_t)BB * HH * SS * DHD;     // 2M f16 elements per tensor
  _Float16* qws  = (_Float16*)d_ws;
  _Float16* kws  = qws + n;
  _Float16* vtws = kws + n;                  // 12 MB total workspace

  proj_kernel<<<(BB * SS) / 16, 256, 0, stream>>>(qx, kx, vx, Wq, bq, Wk, bk, Wv, bv,
                                                  qws, kws, vtws);
  attn_kernel<<<(BB * HH * (SS / 16)) / 8, 256, 0, stream>>>(qws, kws, vtws, amask,
                                                             hmask, (float*)d_out);
}
